// MultiHeadSWCA_45251775430687
// MI455X (gfx1250) — matrix operand order repaired and checked by reading, not compiled, not run on hardware
//
#include <hip/hip_runtime.h>
#include <hip/hip_bf16.h>

#define NHEADS 12
#define DISP 4

typedef __attribute__((ext_vector_type(16))) __bf16 bf16x16;
typedef __attribute__((ext_vector_type(8)))  float  f32x8;

union Frag { bf16x16 v; uint4 q[2]; };

__device__ __forceinline__ unsigned short f2bf(float f) {
    unsigned int u = __builtin_bit_cast(unsigned int, f);
    u += 0x7FFFu + ((u >> 16) & 1u);   // round-to-nearest-even
    return (unsigned short)(u >> 16);
}

// Pack two floats into two bf16 (low = a, high = b).
// Native cvt builtin if present; else round-half-up + one V_PERM_B32.
__device__ __forceinline__ unsigned int pk_bf16(float a, float b) {
#if __has_builtin(__builtin_amdgcn_cvt_pk_bf16_f32)
    typedef __attribute__((ext_vector_type(2))) __bf16 bf16x2;
    bf16x2 r = __builtin_amdgcn_cvt_pk_bf16_f32(a, b);
    return __builtin_bit_cast(unsigned int, r);
#else
    unsigned int ua = __builtin_bit_cast(unsigned int, a) + 0x8000u;
    unsigned int ub = __builtin_bit_cast(unsigned int, b) + 0x8000u;
    return __builtin_amdgcn_perm(ub, ua, 0x07060302u);
#endif
}

// ---------------------------------------------------------------------------
// Pre-kernel (one-shot, tiny): fused logit matrix M = Wq^T*Wk (bf16), wv, wl
// (bf16), t-bias cvec = Wk^T*bq, and 4 variants of (PE bias + masks)*log2(e).
// Loops kept rolled: this runs once on ~4KB and must stay small.
// ---------------------------------------------------------------------------
__global__ void swin_prep_kernel(const float* __restrict__ pe,
                                 const float* __restrict__ wq,
                                 const float* __restrict__ bq,
                                 const float* __restrict__ wk,
                                 const float* __restrict__ wv,
                                 const float* __restrict__ wl,
                                 unsigned short* __restrict__ wbf,
                                 float* __restrict__ tbias,
                                 float* __restrict__ biasvar) {
    int stride = gridDim.x * blockDim.x;
    int t0 = blockIdx.x * blockDim.x + threadIdx.x;
    // wbf[0..1023]: M stored as [e][d] with M[d][e] = sum_f Wq[f][d]*Wk[f][e]
#pragma clang loop unroll(disable)
    for (int idx = t0; idx < 1024; idx += stride) {
        int e = idx >> 5, d = idx & 31;
        float s = 0.0f;
#pragma clang loop unroll(disable)
        for (int f = 0; f < 32; ++f)
            s = fmaf(wq[f * 32 + d], wk[f * 32 + e], s);
        wbf[e * 32 + d] = f2bf(s);
    }
    // wbf[1024..2047] = wv, wbf[2048..3071] = wl (row-major bf16)
#pragma clang loop unroll(disable)
    for (int idx = t0; idx < 2 * 1024; idx += stride) {
        const float* src = (idx < 1024) ? wv : wl;
        wbf[1024 + idx] = f2bf(src[idx & 1023]);
    }
    // tbias[e] = (Wk^T bq)[e]  (surviving per-column cross term)
#pragma clang loop unroll(disable)
    for (int idx = t0; idx < 32; idx += stride) {
        float s = 0.0f;
#pragma clang loop unroll(disable)
        for (int f = 0; f < 32; ++f)
            s = fmaf(bq[f], wk[f * 32 + idx], s);
        tbias[idx] = s;
    }
    // bias variants: [var][64][64] fp32, pre-multiplied by log2(e)
    const float LOG2E = 1.4426950408889634f;
#pragma clang loop unroll(disable)
    for (int idx = t0; idx < 4 * 64 * 64; idx += stride) {
        int var = idx >> 12;
        int ij  = idx & 4095;
        int i = ij >> 6, j = ij & 63;
        int ri = (j >> 3) - (i >> 3) + 7;
        int rj = (j & 7)  - (i & 7)  + 7;
        float v = pe[ri * 15 + rj];
        if (var & 1) { if (((i >= 32) ? 1 : 0) != ((j >= 32) ? 1 : 0)) v -= 1.0e9f; }
        if (var & 2) { if ((((i & 7) >= 4) ? 1 : 0) != (((j & 7) >= 4) ? 1 : 0)) v -= 1.0e9f; }
        biasvar[idx] = v * LOG2E;
    }
}

// ---------------------------------------------------------------------------
// Main kernel: one block (4 waves) per (batch, head, window).
// Wave w owns rows [16w, 16w+16) of the 64-token window.
// ---------------------------------------------------------------------------
__global__ __launch_bounds__(128) void swin_attn_kernel(
        const float* __restrict__ feat,
        const float* __restrict__ bvb, const float* __restrict__ blb,
        const unsigned short* __restrict__ wbf,
        const float* __restrict__ tbias,
        const float* __restrict__ biasvar,
        float* __restrict__ out) {
    // padded rows to de-conflict LDS banks; fragment loads stay 16B-aligned
    __shared__ unsigned short xs[64][40];   // x window (bf16); B-side of logits
    __shared__ unsigned short ts[64][40];   // t = x@M + cvec (A-side of logits)
    __shared__ unsigned short os[64][40];   // unnormalized Y = P@v staging
    __shared__ unsigned short vsT[32][72];  // v transposed: vsT[d][j]
    __shared__ unsigned short ps[64][72];   // unnormalized softmax probs (bf16)

    const int blk = blockIdx.x;
    const int n   = blk & 63;
    const int h   = (blk >> 6) % NHEADS;
    const int b   = blk / (64 * NHEADS);
    const int wy  = n >> 3, wx = n & 7;
    const int variant = ((wy == 7) ? 1 : 0) | ((wx == 7) ? 2 : 0);

    const int tid  = threadIdx.x;
    const int lane = tid & 31;
    const int wave = tid >> 5;
    const int lr   = lane & 15;   // N (or M) index within a tile
    const int lh   = lane >> 4;   // half-wave selector

    // ---- Phase 1: gather shifted window -> LDS bf16 (row-major 64x32) ----
    {
        int r  = tid >> 1;
        int c0 = (tid & 1) * 16;
        int sy = r >> 3, sx = r & 7;
        int gy = (wy * 8 + sy + DISP) & 63;
        int gx = (wx * 8 + sx + DISP) & 63;
        const float* src = feat + (((size_t)b * 4096 + (size_t)(gy * 64 + gx)) * 384
                                   + h * 32 + c0);
        float4 f0 = ((const float4*)src)[0];
        float4 f1 = ((const float4*)src)[1];
        float4 f2 = ((const float4*)src)[2];
        float4 f3 = ((const float4*)src)[3];
        uint4 p0, p1;
        p0.x = pk_bf16(f0.x, f0.y);
        p0.y = pk_bf16(f0.z, f0.w);
        p0.z = pk_bf16(f1.x, f1.y);
        p0.w = pk_bf16(f1.z, f1.w);
        p1.x = pk_bf16(f2.x, f2.y);
        p1.y = pk_bf16(f2.z, f2.w);
        p1.z = pk_bf16(f3.x, f3.y);
        p1.w = pk_bf16(f3.z, f3.w);
        *(uint4*)&xs[r][c0]     = p0;
        *(uint4*)&xs[r][c0 + 8] = p1;
    }
    __syncthreads();

    // ---- Phase 2: t = x@M + cvec ; v = x@wv^T + bv  (4 WMMAs/wave) ----
    Frag ax;
    {
        int row = wave * 16 + lr;
        ax.q[0] = *(const uint4*)&xs[row][lh * 8];
        ax.q[1] = *(const uint4*)&xs[row][16 + lh * 8];
    }
#pragma unroll
    for (int p = 0; p < 2; ++p) {
        const unsigned short* wmat = wbf + p * 1024;   // p=0: M, p=1: wv
#pragma unroll
        for (int nt = 0; nt < 2; ++nt) {
            Frag bw;
            bw.q[0] = *(const uint4*)&wmat[(nt * 16 + lr) * 32 + lh * 8];
            bw.q[1] = *(const uint4*)&wmat[(nt * 16 + lr) * 32 + 16 + lh * 8];
            f32x8 acc = {};
            acc = __builtin_amdgcn_wmma_f32_16x16x32_bf16(
                      false, ax.v, false, bw.v, (short)0, acc, false, false);
            float bj = (p == 0) ? tbias[nt * 16 + lr] : bvb[nt * 16 + lr];
            // vector splat-add: lowers to v_pk_add_f32 on the natural WMMA
            // dest pairs, avoiding SLP's cross-pair register shuffling
            f32x8 sb = acc + bj;
            int col  = nt * 16 + lr;
            int rb   = wave * 16 + lh * 8;
#pragma unroll
            for (int g = 0; g < 8; g += 2) {
                unsigned int pk = pk_bf16(sb[g], sb[g + 1]);
                if (p == 0) {
                    ts[rb + g][col]     = (unsigned short)pk;
                    ts[rb + g + 1][col] = (unsigned short)(pk >> 16);
                } else {
                    // transposed staging: rows contiguous -> one b32 store
                    *(unsigned int*)&vsT[col][rb + g] = pk;
                }
            }
        }
    }
    __syncthreads();

    // ---- Phase 3: logits = t @ x^T  (4 WMMAs/wave; B-side = raw x rows) ----
    f32x8 accq[4];
    {
        Frag aq;
        int row = wave * 16 + lr;
        aq.q[0] = *(const uint4*)&ts[row][lh * 8];
        aq.q[1] = *(const uint4*)&ts[row][16 + lh * 8];
#pragma unroll
        for (int nt = 0; nt < 4; ++nt) {
            Frag bxf;
            bxf.q[0] = *(const uint4*)&xs[nt * 16 + lr][lh * 8];
            bxf.q[1] = *(const uint4*)&xs[nt * 16 + lr][16 + lh * 8];
            f32x8 z = {};
            accq[nt] = __builtin_amdgcn_wmma_f32_16x16x32_bf16(
                           false, aq.v, false, bxf.v, (short)0, z, false, false);
        }
    }

    // ---- Phase 4: probs = exp2(logit*scale + bias)  (log2-domain) ----
    // Logits are bounded here, so max-subtraction is numerically inert and
    // omitted; per-row terms dropped from the fused logits cancel exactly in
    // softmax. Masked entries carry -1.44e9 -> exp2 -> exactly 0.
    const float SCL2 = 0.25506600286896974f;    // (1/sqrt(32)) * log2(e)
    const float* bvt = biasvar + variant * 4096;
    const int ib = wave * 16 + lh * 8;
#pragma unroll
    for (int nt = 0; nt < 4; ++nt)
#pragma unroll
        for (int g = 0; g < 8; ++g)
            accq[nt][g] = exp2f(fmaf(accq[nt][g], SCL2,
                                     bvt[(ib + g) * 64 + nt * 16 + lr]));
    // store unnormalized probs (pairs g,g+1 -> b16 lo/hi stores)
#pragma unroll
    for (int g = 0; g < 8; g += 2)
#pragma unroll
        for (int nt = 0; nt < 4; ++nt) {
            unsigned int pk = pk_bf16(accq[nt][g], accq[nt][g + 1]);
            ps[ib + g][nt * 16 + lr]     = (unsigned short)pk;
            ps[ib + g + 1][nt * 16 + lr] = (unsigned short)(pk >> 16);
        }

    // ---- Phase 5: Y = P @ v and row-sums via ones-WMMA (6 WMMAs/wave) ----
    Frag ones;
    ones.q[0] = make_uint4(0x3F803F80u, 0x3F803F80u, 0x3F803F80u, 0x3F803F80u);
    ones.q[1] = ones.q[0];
    f32x8 acco[2] = {{}, {}};
    f32x8 accs = {};
#pragma unroll
    for (int kt = 0; kt < 2; ++kt) {
        Frag ap;
        int row = wave * 16 + lr;
        ap.q[0] = *(const uint4*)&ps[row][kt * 32 + lh * 8];
        ap.q[1] = *(const uint4*)&ps[row][kt * 32 + 16 + lh * 8];
#pragma unroll
        for (int nt = 0; nt < 2; ++nt) {
            Frag bvf;
            bvf.q[0] = *(const uint4*)&vsT[nt * 16 + lr][kt * 32 + lh * 8];
            bvf.q[1] = *(const uint4*)&vsT[nt * 16 + lr][kt * 32 + 16 + lh * 8];
            acco[nt] = __builtin_amdgcn_wmma_f32_16x16x32_bf16(
                           false, ap.v, false, bvf.v, (short)0, acco[nt], false, false);
        }
        // row sums: P @ ones -> every lane of a row holds the row's sum
        accs = __builtin_amdgcn_wmma_f32_16x16x32_bf16(
                   false, ap.v, false, ones.v, (short)0, accs, false, false);
    }
    float rs[8];                               // per-row 1/sum, applied at the end
#pragma unroll
    for (int g = 0; g < 8; ++g)
        rs[g] = __builtin_amdgcn_rcpf(accs[g]);
    // stage UNNORMALIZED Y; normalization commutes with @ wl^T
#pragma unroll
    for (int nt = 0; nt < 2; ++nt)
#pragma unroll
        for (int g = 0; g < 8; g += 2) {
            unsigned int pk = pk_bf16(acco[nt][g], acco[nt][g + 1]);
            os[ib + g][nt * 16 + lr]     = (unsigned short)pk;
            os[ib + g + 1][nt * 16 + lr] = (unsigned short)(pk >> 16);
        }

    // ---- Phase 6: out = diag(rs) * (Y @ wl^T) + bl (2 WMMAs/wave) ----
    Frag ao;
    {
        int row = wave * 16 + lr;
        ao.q[0] = *(const uint4*)&os[row][lh * 8];
        ao.q[1] = *(const uint4*)&os[row][16 + lh * 8];
    }
    const unsigned short* wlb = wbf + 2 * 1024;
    f32x8 acc2[2];
#pragma unroll
    for (int nt = 0; nt < 2; ++nt) {
        Frag bw;
        bw.q[0] = *(const uint4*)&wlb[(nt * 16 + lr) * 32 + lh * 8];
        bw.q[1] = *(const uint4*)&wlb[(nt * 16 + lr) * 32 + 16 + lh * 8];
        f32x8 z = {};
        acc2[nt] = __builtin_amdgcn_wmma_f32_16x16x32_bf16(
                       false, ao.v, false, bw.v, (short)0, z, false, false);
    }
    {
        float bj0 = blb[lr];
        float bj1 = blb[16 + lr];
#pragma unroll
        for (int g = 0; g < 8; ++g) {
            int i  = ib + g;
            int sy = i >> 3, sx = i & 7;
            int gy = (wy * 8 + sy + DISP) & 63;   // inverse cyclic shift
            int gx = (wx * 8 + sx + DISP) & 63;
            float* row = out + ((size_t)b * 4096 + (size_t)(gy * 64 + gx)) * 384
                             + h * 32;
            row[lr]      = fmaf(acc2[0][g], rs[g], bj0);
            row[16 + lr] = fmaf(acc2[1][g], rs[g], bj1);
        }
    }
}

extern "C" void kernel_launch(void* const* d_in, const int* in_sizes, int n_in,
                              void* d_out, int out_size, void* d_ws, size_t ws_size,
                              hipStream_t stream) {
    (void)n_in; (void)out_size; (void)ws_size;
    const float* feat = (const float*)d_in[0];
    const float* pe   = (const float*)d_in[1];
    const float* wq   = (const float*)d_in[2];
    const float* bq   = (const float*)d_in[3];
    const float* wk   = (const float*)d_in[4];
    const float* wv   = (const float*)d_in[6];
    const float* bv   = (const float*)d_in[7];
    const float* wl   = (const float*)d_in[8];
    const float* bl   = (const float*)d_in[9];

    int Bsz = in_sizes[0] / (4096 * 384);   // 32

    unsigned short* wbf  = (unsigned short*)d_ws;          // 3*1024 bf16 = 6 KB
    float*          tb   = (float*)((char*)d_ws + 6144);   // 32 floats
    float*          bvar = (float*)((char*)d_ws + 8192);   // 64 KB

    swin_prep_kernel<<<16, 256, 0, stream>>>(pe, wq, bq, wk, wv, wl,
                                             wbf, tb, bvar);

    int nblocks = Bsz * NHEADS * 64;
    swin_attn_kernel<<<nblocks, 128, 0, stream>>>(feat, bv, bl,
                                                  wbf, tb, bvar, (float*)d_out);
}
